// Cross_Modality_Relevance_82721070121286
// MI455X (gfx1250) — matrix operand order repaired and checked by reading, not compiled, not run on hardware
//
#include <hip/hip_runtime.h>
#include <math.h>

// ---------------------------------------------------------------------------
// Types for CDNA5 WMMA
// ---------------------------------------------------------------------------
typedef __attribute__((ext_vector_type(16))) _Float16 v16h;
typedef __attribute__((ext_vector_type(8)))  _Float16 v8h;
typedef __attribute__((ext_vector_type(8)))  float    v8f;
typedef __attribute__((ext_vector_type(4)))  float    f4;

enum { AM_PLAIN = 0, AM_LANGPAIR = 1, AM_IMGPAIR = 2 };
enum { ACT_NONE = 0, ACT_RELU = 1, ACT_GELU = 2 };

#define NEG_BIG (-3.402823466e38f)

// ---------------------------------------------------------------------------
// Activation epilogue
// ---------------------------------------------------------------------------
template <int ACT>
__device__ __forceinline__ float apply_act(float x) {
  if (ACT == ACT_RELU) return x > 0.f ? x : 0.f;
  if (ACT == ACT_GELU) return 0.5f * x * (1.0f + erff(x * 0.70710678118654752f));
  return x;
}

// ---------------------------------------------------------------------------
// WMMA GEMM:  C(M,N) = act( A(M,K) @ W(K,N) + bias(N) )
//
//   Block tile 64x64, BK = 64 (two 16x16x32 K-steps per stage), 256 thr = 8 waves.
//   Wave w: rows [16*(w&3),+16), cols [32*(w>>2),+32): 1 A frag + 2 B frags +
//   2 wmma per K-step -> 4 v_wmma_f32_16x16x32_f16 per stage.
//
//   LDS tiles are stored in WMMA *fragment layout*:
//     As2[ks][rowTile][lane][16 halfs], Bs2[ks][colTile][lane][16 halfs]
//   so a fragment is one aligned 32B v16h load and staging writes are aligned
//   16B v8h stores (ds_store_b128), no scalar u16 traffic.
//
//   A-operand modes: plain row-major / lang pair-concat / img pair-add.
//   Pair tril-table indices are resolved ONCE per thread before the K loop.
//   Requires K % 32 == 0 (true for every layer here); K % 64 == 32 tails are
//   zero-filled.  M and N are guarded.
// ---------------------------------------------------------------------------
template <int AMODE, int ACT>
__global__ __launch_bounds__(256) void gemm_wmma(
    const float* __restrict__ A, const float* __restrict__ W,
    const float* __restrict__ bias, float* __restrict__ Cout,
    int M, int N, int K, int lda,
    const int* __restrict__ pr, const int* __restrict__ pc)
{
  __shared__ __attribute__((aligned(64))) _Float16 As2[4096];
  __shared__ __attribute__((aligned(64))) _Float16 Bs2[4096];

  const int tid  = threadIdx.x;
  const int lane = tid & 31;
  const int wave = tid >> 5;
  const int wm   = wave & 3;   // 16-row tile
  const int wn   = wave >> 2;  // 32-col group
  const int rowBase = blockIdx.y * 64;
  const int colBase = blockIdx.x * 64;

  // ---- per-thread A staging task: row am, 16-wide k chunk aq ----
  const int am = tid >> 2;          // 0..63
  const int aq = tid & 3;           // 0..3
  const int arow = rowBase + am;
  const bool arowOK = (arow < M);
  const float* aptr0;
  const float* aptr1 = nullptr;
  {
    int r = arowOK ? arow : 0;
    if (AMODE == AM_PLAIN) {
      aptr0 = A + (size_t)r * lda;
    } else if (AMODE == AM_LANGPAIR) {
      int b = r / 190, p = r - b * 190;
      aptr0 = A + ((size_t)b * 20 + pr[p]) * 768;   // k < 768
      aptr1 = A + ((size_t)b * 20 + pc[p]) * 768;   // k >= 768 (use k-768)
    } else {
      int b = r / 630, p = r - b * 630;
      aptr0 = A + ((size_t)b * 36 + pr[p]) * 768;
      aptr1 = A + ((size_t)b * 36 + pc[p]) * 768;
    }
  }
  // A fragment-layout destinations (two aligned v8h slots)
  const int aks = aq >> 1;
  _Float16* adst_lo = As2 + (((aks * 4 + (am >> 4)) * 32 + (am & 15)) * 16 + (aq & 1) * 8);
  _Float16* adst_hi = adst_lo + 256;   // lane + 16

  // ---- per-thread B staging task: col bn, 16-wide k chunk bq ----
  const int bn = tid & 63;
  const int bq = tid >> 6;          // 0..3
  const int bks = bq >> 1;
  const int bcol = colBase + bn;
  const bool bcolOK = (bcol < N);
  _Float16* bdst = Bs2 + (((bks * 4 + (bn >> 4)) * 32 + ((bn & 15) + 16 * (bq & 1))) * 16);
  const float* wbase0 = W + bcol;   // + k*N

  v8f acc0 = {}, acc1 = {};

  for (int k0 = 0; k0 < K; k0 += 64) {
    // ---- stage A (16 floats -> 2 x v8h) ----
    {
      const int kq = k0 + 16 * aq;
      float v[16];
      if (arowOK && kq < K) {
        if (AMODE == AM_IMGPAIR) {
          const f4* s0 = (const f4*)(aptr0 + kq);
          const f4* s1 = (const f4*)(aptr1 + kq);
#pragma unroll
          for (int t = 0; t < 4; ++t) {
            f4 x = s0[t] + s1[t];
            v[4 * t + 0] = x[0]; v[4 * t + 1] = x[1]; v[4 * t + 2] = x[2]; v[4 * t + 3] = x[3];
          }
        } else {
          const float* base;
          if (AMODE == AM_LANGPAIR) base = (kq < 768) ? (aptr0 + kq) : (aptr1 + (kq - 768));
          else                      base = aptr0 + kq;
          const f4* s = (const f4*)base;
#pragma unroll
          for (int t = 0; t < 4; ++t) {
            f4 x = s[t];
            v[4 * t + 0] = x[0]; v[4 * t + 1] = x[1]; v[4 * t + 2] = x[2]; v[4 * t + 3] = x[3];
          }
        }
      } else {
#pragma unroll
        for (int j = 0; j < 16; ++j) v[j] = 0.f;
      }
      v8h lo, hi;
#pragma unroll
      for (int j = 0; j < 8; ++j) { lo[j] = (_Float16)v[j]; hi[j] = (_Float16)v[j + 8]; }
      *(v8h*)adst_lo = lo;
      *(v8h*)adst_hi = hi;
    }
    // ---- stage B (16 floats down column -> 2 x v8h) ----
    {
      const int kq = k0 + 32 * bks + 8 * (bq & 1);
      float v[16];
      if (bcolOK && kq < K) {
        const float* s = wbase0 + (size_t)kq * N;
#pragma unroll
        for (int j = 0; j < 16; ++j) v[j] = s[(size_t)(j + (j & 8)) * N];
      } else {
#pragma unroll
        for (int j = 0; j < 16; ++j) v[j] = 0.f;
      }
      v8h lo, hi;
#pragma unroll
      for (int j = 0; j < 8; ++j) { lo[j] = (_Float16)v[j]; hi[j] = (_Float16)v[j + 8]; }
      *(v8h*)bdst = lo;
      *(v8h*)(bdst + 8) = hi;
    }
    __syncthreads();

#pragma unroll
    for (int ks = 0; ks < 2; ++ks) {
      const v16h a  = *(const v16h*)(As2 + ((ks * 4 + wm) * 32 + lane) * 16);
      const v16h b0 = *(const v16h*)(Bs2 + ((ks * 4 + 2 * wn) * 32 + lane) * 16);
      const v16h b1 = *(const v16h*)(Bs2 + ((ks * 4 + 2 * wn + 1) * 32 + lane) * 16);
      acc0 = __builtin_amdgcn_wmma_f32_16x16x32_f16(false, a, false, b0,
                                                    (short)0, acc0, false, false);
      acc1 = __builtin_amdgcn_wmma_f32_16x16x32_f16(false, a, false, b1,
                                                    (short)0, acc1, false, false);
    }
    __syncthreads();
  }

  // epilogue: C/D layout -> lane holds N = lane&15 (+16 frag1), M = r + 8*(lane>>4)
  const int g   = lane >> 4;
  const int mL  = (wm << 4) + (g << 3);
  const int nL0 = colBase + (wn << 5) + (lane & 15);
#pragma unroll
  for (int r = 0; r < 8; ++r) {
    int row = rowBase + mL + r;
    if (row < M) {
      if (nL0 < N)      Cout[(size_t)row * N + nL0]      = apply_act<ACT>(acc0[r] + bias[nL0]);
      if (nL0 + 16 < N) Cout[(size_t)row * N + nL0 + 16] = apply_act<ACT>(acc1[r] + bias[nL0 + 16]);
    }
  }
}

// ---------------------------------------------------------------------------
// tril_indices(n, -1) tables
// ---------------------------------------------------------------------------
__device__ __forceinline__ void tril_rc(int p, int* r, int* c) {
  int rr = (int)((1.0f + sqrtf(1.0f + 8.0f * (float)p)) * 0.5f);
  while (rr * (rr - 1) / 2 > p) --rr;
  while ((rr + 1) * rr / 2 <= p) ++rr;
  *r = rr;
  *c = p - rr * (rr - 1) / 2;
}

__global__ void init_pairs_k(int* rl, int* cl, int* ri, int* ci) {
  for (int p = threadIdx.x; p < 190; p += 256) { int r, c; tril_rc(p, &r, &c); rl[p] = r; cl[p] = c; }
  for (int p = threadIdx.x; p < 630; p += 256) { int r, c; tril_rc(p, &r, &c); ri[p] = r; ci[p] = c; }
}

// ---------------------------------------------------------------------------
// out[b,l,r] = <x_bl, y_br> / (max(|x_bl|,eps) * max(|y_br|,eps)); one wave/dot
// ---------------------------------------------------------------------------
__global__ __launch_bounds__(256) void normdot_k(
    const float* __restrict__ X, int bsx, int lsx, int nl,
    const float* __restrict__ Y, int bsy, int lsy, int nr,
    int d, int nb, float* __restrict__ out)
{
  int gtid = blockIdx.x * 256 + threadIdx.x;
  int wid = gtid >> 5, lane = gtid & 31;
  int total = nb * nl * nr;
  if (wid >= total) return;
  int r = wid % nr;
  int t = wid / nr;
  int l = t % nl;
  int b = t / nl;
  const float* xp = X + (size_t)b * bsx + (size_t)l * lsx;
  const float* yp = Y + (size_t)b * bsy + (size_t)r * lsy;
  float sxy = 0.f, sxx = 0.f, syy = 0.f;
  for (int k = lane; k < d; k += 32) {
    float xv = xp[k], yv = yp[k];
    sxy += xv * yv; sxx += xv * xv; syy += yv * yv;
  }
  for (int off = 16; off > 0; off >>= 1) {
    sxy += __shfl_xor(sxy, off, 32);
    sxx += __shfl_xor(sxx, off, 32);
    syy += __shfl_xor(syy, off, 32);
  }
  if (lane == 0) {
    float nx = fmaxf(sqrtf(sxx), 1e-12f);
    float ny = fmaxf(sqrtf(syy), 1e-12f);
    out[wid] = sxy / (nx * ny);
  }
}

// ---------------------------------------------------------------------------
// 3x3 VALID conv + ReLU + 2x2 maxpool (stride 2, VALID)
// ---------------------------------------------------------------------------
__global__ __launch_bounds__(256) void conv3_relu_pool_k(
    const float* __restrict__ x, const float* __restrict__ w,
    const float* __restrict__ bias, float* __restrict__ y,
    int N, int Cin, int Hin, int Win, int Cout)
{
  int Hc = Hin - 2, Wc = Win - 2, Hp = Hc >> 1, Wp = Wc >> 1;
  int total = N * Cout * Hp * Wp;
  for (int idx = blockIdx.x * 256 + threadIdx.x; idx < total; idx += gridDim.x * 256) {
    int pw = idx % Wp; int t = idx / Wp;
    int ph = t % Hp;   t /= Hp;
    int co = t % Cout; int n = t / Cout;
    float best = NEG_BIG;
    for (int dy = 0; dy < 2; ++dy)
      for (int dx = 0; dx < 2; ++dx) {
        int oh = (ph << 1) + dy, ow = (pw << 1) + dx;
        float acc = bias[co];
        for (int ci = 0; ci < Cin; ++ci) {
          const float* xp = x + ((size_t)(n * Cin + ci) * Hin + oh) * Win + ow;
          const float* wp = w + (size_t)(co * Cin + ci) * 9;
          acc += xp[0] * wp[0] + xp[1] * wp[1] + xp[2] * wp[2]
               + xp[Win] * wp[3] + xp[Win + 1] * wp[4] + xp[Win + 2] * wp[5]
               + xp[2 * Win] * wp[6] + xp[2 * Win + 1] * wp[7] + xp[2 * Win + 2] * wp[8];
        }
        acc = acc > 0.f ? acc : 0.f;
        best = best > acc ? best : acc;
      }
    y[idx] = best;
  }
}

// ---------------------------------------------------------------------------
// top-10 (descending, ties -> lower index) per batch row; one wave per row
// ---------------------------------------------------------------------------
__global__ __launch_bounds__(32) void topk_k(const float* __restrict__ score,
                                             int P, int* __restrict__ outIdx)
{
  __shared__ float s[640];
  int b = blockIdx.x, lane = threadIdx.x;
  for (int i = lane; i < P; i += 32) s[i] = score[(size_t)b * P + i];
  __syncthreads();
  for (int k = 0; k < 10; ++k) {
    float bv = NEG_BIG;
    int   bi = 0x7fffffff;
    for (int i = lane; i < P; i += 32) {
      float v = s[i];
      if (v > bv || (v == bv && i < bi)) { bv = v; bi = i; }
    }
    for (int off = 16; off > 0; off >>= 1) {
      float ov = __shfl_xor(bv, off, 32);
      int   oi = __shfl_xor(bi, off, 32);
      if (ov > bv || (ov == bv && oi < bi)) { bv = ov; bi = oi; }
    }
    if (lane == 0) {
      outIdx[b * 10 + k] = bi;
      if (bi >= 0 && bi < P) s[bi] = NEG_BIG;
    }
    __syncthreads();
  }
}

// ---------------------------------------------------------------------------
// gathers
// ---------------------------------------------------------------------------
__global__ void gather_lang_k(const float* __restrict__ C, const int* __restrict__ li,
                              float* __restrict__ out) {
  int idx = blockIdx.x * 256 + threadIdx.x;
  if (idx >= 64 * 10 * 768) return;
  int k = idx % 768; int t = idx / 768; int j = t % 10; int b = t / 10;
  out[idx] = C[((size_t)b * 190 + li[b * 10 + j]) * 768 + k];
}

__global__ void gather_imgpair_k(const float* __restrict__ img, const int* __restrict__ ii,
                                 const int* __restrict__ ri, const int* __restrict__ ci,
                                 float* __restrict__ out) {
  int idx = blockIdx.x * 256 + threadIdx.x;
  if (idx >= 64 * 10 * 768) return;
  int k = idx % 768; int t = idx / 768; int j = t % 10; int b = t / 10;
  int p = ii[b * 10 + j];
  size_t base = (size_t)b * 36 * 768 + k;
  out[idx] = img[base + (size_t)ri[p] * 768] + img[base + (size_t)ci[p] * 768];
}

// ---------------------------------------------------------------------------
// layernorm over last dim W (eps = 1e-12, biased variance); one block per row
// ---------------------------------------------------------------------------
__global__ __launch_bounds__(256) void layernorm_k(const float* __restrict__ x,
                                                   const float* __restrict__ g,
                                                   const float* __restrict__ b,
                                                   float* __restrict__ y, int W)
{
  __shared__ float red[256];
  int row = blockIdx.x;
  const float* xr = x + (size_t)row * W;
  float* yr = y + (size_t)row * W;

  float s = 0.f;
  for (int i = threadIdx.x; i < W; i += 256) s += xr[i];
  red[threadIdx.x] = s;
  __syncthreads();
  for (int st = 128; st > 0; st >>= 1) {
    if (threadIdx.x < st) red[threadIdx.x] += red[threadIdx.x + st];
    __syncthreads();
  }
  float mean = red[0] / (float)W;
  __syncthreads();

  float v = 0.f;
  for (int i = threadIdx.x; i < W; i += 256) { float d = xr[i] - mean; v += d * d; }
  red[threadIdx.x] = v;
  __syncthreads();
  for (int st = 128; st > 0; st >>= 1) {
    if (threadIdx.x < st) red[threadIdx.x] += red[threadIdx.x + st];
    __syncthreads();
  }
  float inv = rsqrtf(red[0] / (float)W + 1e-12f);

  for (int i = threadIdx.x; i < W; i += 256)
    yr[i] = (xr[i] - mean) * inv * g[i] + b[i];
}

// ---------------------------------------------------------------------------
// concat 4 x (32,768) -> (32,3072)
// ---------------------------------------------------------------------------
__global__ void concat4_k(const float* __restrict__ a, const float* __restrict__ b,
                          const float* __restrict__ c, const float* __restrict__ d,
                          float* __restrict__ out) {
  int idx = blockIdx.x * 256 + threadIdx.x;
  if (idx >= 32 * 3072) return;
  int col = idx % 3072, row = idx / 3072;
  float v;
  if (col < 768)       v = a[row * 768 + col];
  else if (col < 1536) v = b[row * 768 + col - 768];
  else if (col < 2304) v = c[row * 768 + col - 1536];
  else                 v = d[row * 768 + col - 2304];
  out[idx] = v;
}

// ---------------------------------------------------------------------------
// host-side dispatch
// ---------------------------------------------------------------------------
static void gemm(hipStream_t s, int amode, int act, const float* A, const float* W,
                 const float* b, float* C, int M, int N, int K, int lda,
                 const int* pr = nullptr, const int* pc = nullptr)
{
  dim3 grid((unsigned)((N + 63) / 64), (unsigned)((M + 63) / 64)), block(256);
  if (amode == AM_LANGPAIR)
    gemm_wmma<AM_LANGPAIR, ACT_RELU><<<grid, block, 0, s>>>(A, W, b, C, M, N, K, lda, pr, pc);
  else if (amode == AM_IMGPAIR)
    gemm_wmma<AM_IMGPAIR, ACT_RELU><<<grid, block, 0, s>>>(A, W, b, C, M, N, K, lda, pr, pc);
  else if (act == ACT_RELU)
    gemm_wmma<AM_PLAIN, ACT_RELU><<<grid, block, 0, s>>>(A, W, b, C, M, N, K, lda, pr, pc);
  else if (act == ACT_GELU)
    gemm_wmma<AM_PLAIN, ACT_GELU><<<grid, block, 0, s>>>(A, W, b, C, M, N, K, lda, pr, pc);
  else
    gemm_wmma<AM_PLAIN, ACT_NONE><<<grid, block, 0, s>>>(A, W, b, C, M, N, K, lda, pr, pc);
}

static inline unsigned blocks_for(long long total) {
  return (unsigned)((total + 255) / 256);
}

extern "C" void kernel_launch(void* const* d_in, const int* in_sizes, int n_in,
                              void* d_out, int out_size, void* d_ws, size_t ws_size,
                              hipStream_t stream) {
  (void)in_sizes; (void)n_in; (void)out_size; (void)ws_size;

  const float* lang  = (const float*)d_in[0];   // (64,20,768)
  const float* img   = (const float*)d_in[1];   // (64,36,768)
  const float* cross = (const float*)d_in[2];   // (64,768) -> viewed (32,1536)
#define PW(i) ((const float*)d_in[i])

  // ---- workspace bump allocator (units: floats; peak ~178 MB) ----
  float* ws = (float*)d_ws;
  size_t off = 0;
  auto alloc = [&](size_t n) -> float* {
    float* p = ws + off;
    off += (n + 63) & ~(size_t)63;
    return p;
  };

  float* big0     = alloc(18677760);  // h_big (12160x1536), reused for irel_h1
  float* Cbuf     = alloc(9338880);   // lang_pair (12160x768)
  float* lrel1    = alloc(4669440);   // 12160x384
  float* lrel2    = alloc(2334720);   // 12160x192
  float* lscore   = alloc(12160);
  float* irel2    = alloc(7741440);   // 40320x192
  float* iscore   = alloc(40320);
  int*   li       = (int*)alloc(640);
  int*   ii       = (int*)alloc(640);
  int*   prl      = (int*)alloc(190);
  int*   pcl      = (int*)alloc(190);
  int*   pri      = (int*)alloc(630);
  int*   pci      = (int*)alloc(630);
  float* lang_rel = alloc(491520);    // (64,10,768)
  float* img_rel  = alloc(491520);
  float* rc       = alloc(6400);      // (64,10,10)
  float* h4       = alloc(32768);     // (64,32,4,4)
  float* relate   = alloc(49152);     // (64,768) = (32,1536)
  float* fc4_h    = alloc(49152);
  float* fc4_ln   = alloc(49152);
  float* logit4   = alloc(24576);
  float* c12      = alloc(4608);      // (32,12,12)
  float* hc       = alloc(25600);     // (32,32,5,5)
  float* cross_fc = alloc(24576);
  float* fc1_h    = alloc(24576);
  float* logit1   = alloc(24576);
  float* cis      = alloc(46080);     // (64,20,36)
  float* he1      = alloc(156672);    // (64,16,9,17)
  float* he2      = alloc(43008);     // (64,32,3,7)
  float* ent_fc   = alloc(49152);     // (64,768) = (32,1536)
  float* fc2_h    = alloc(49152);
  float* fc2_ln   = alloc(49152);
  float* logit2   = alloc(24576);
  float* cii      = alloc(41472);     // (32,36,36)
  float* hi1      = alloc(147968);    // (32,16,17,17)
  float* hi2      = alloc(50176);     // (32,32,7,7)
  float* img_fc   = alloc(24576);
  float* fc3_h    = alloc(24576);
  float* logit3   = alloc(24576);
  float* clb      = alloc(98304);     // (32,3072)
  float* f_h      = alloc(98304);
  float* f_ln     = alloc(98304);
  float* f2       = alloc(24576);
  float* f3       = alloc(6144);
  float* irel1    = big0;             // 40320x384 fits in 12160x1536 region

  // ---- pair index tables ----
  init_pairs_k<<<1, 256, 0, stream>>>(prl, pcl, pri, pci);

  // ---- lang pair MLP (pair-concat fused into A operand) ----
  gemm(stream, AM_LANGPAIR, ACT_RELU, lang, PW(3), PW(4), big0, 12160, 1536, 1536, 0, prl, pcl);
  gemm(stream, AM_PLAIN, ACT_NONE, big0, PW(5), PW(6), Cbuf, 12160, 768, 1536, 1536);

  // ---- lang relevance MLP -> score ----
  gemm(stream, AM_PLAIN, ACT_RELU, Cbuf,  PW(7),  PW(8),  lrel1,  12160, 384, 768, 768);
  gemm(stream, AM_PLAIN, ACT_RELU, lrel1, PW(9),  PW(10), lrel2,  12160, 192, 384, 384);
  gemm(stream, AM_PLAIN, ACT_NONE, lrel2, PW(11), PW(12), lscore, 12160, 1,   192, 192);

  // ---- img relevance MLP (pair-add fused into A operand) -> score ----
  gemm(stream, AM_IMGPAIR, ACT_RELU, img, PW(13), PW(14), irel1, 40320, 384, 768, 0, pri, pci);
  gemm(stream, AM_PLAIN, ACT_RELU, irel1, PW(15), PW(16), irel2,  40320, 192, 384, 384);
  gemm(stream, AM_PLAIN, ACT_NONE, irel2, PW(17), PW(18), iscore, 40320, 1,   192, 192);

  // ---- top-k + gathers ----
  topk_k<<<64, 32, 0, stream>>>(lscore, 190, li);
  topk_k<<<64, 32, 0, stream>>>(iscore, 630, ii);
  gather_lang_k<<<blocks_for(64 * 10 * 768), 256, 0, stream>>>(Cbuf, li, lang_rel);
  gather_imgpair_k<<<blocks_for(64 * 10 * 768), 256, 0, stream>>>(img, ii, pri, pci, img_rel);

  // ---- relate branch: normdot -> conv -> fc -> gelu/ln -> logit4 ----
  normdot_k<<<blocks_for((long long)64 * 10 * 10 * 32), 256, 0, stream>>>(
      lang_rel, 7680, 768, 10, img_rel, 7680, 768, 10, 768, 64, rc);
  conv3_relu_pool_k<<<blocks_for(64 * 32 * 4 * 4), 256, 0, stream>>>(
      rc, PW(19), PW(20), h4, 64, 1, 10, 10, 32);
  gemm(stream, AM_PLAIN, ACT_RELU, h4, PW(21), PW(22), relate, 64, 768, 512, 512);
  gemm(stream, AM_PLAIN, ACT_GELU, relate, PW(23), PW(24), fc4_h, 32, 1536, 1536, 1536);
  layernorm_k<<<32, 256, 0, stream>>>(fc4_h, PW(25), PW(26), fc4_ln, 1536);
  gemm(stream, AM_PLAIN, ACT_NONE, fc4_ln, PW(27), PW(28), logit4, 32, 768, 1536, 1536);

  // ---- cross branch -> logit1 ----
  normdot_k<<<blocks_for((long long)32 * 12 * 12 * 32), 256, 0, stream>>>(
      cross, 1536, 64, 12, cross + 768, 1536, 64, 12, 64, 32, c12);
  conv3_relu_pool_k<<<blocks_for(32 * 32 * 5 * 5), 256, 0, stream>>>(
      c12, PW(29), PW(30), hc, 32, 1, 12, 12, 32);
  gemm(stream, AM_PLAIN, ACT_RELU, hc, PW(31), PW(32), cross_fc, 32, 768, 800, 800);
  gemm(stream, AM_PLAIN, ACT_GELU, cross_fc, PW(33), PW(34), fc1_h, 32, 768, 768, 768);
  layernorm_k<<<32, 256, 0, stream>>>(fc1_h, PW(35), PW(36), logit1, 768);

  // ---- lang-img entity branch -> logit2 ----
  normdot_k<<<blocks_for((long long)64 * 20 * 36 * 32), 256, 0, stream>>>(
      lang, 15360, 768, 20, img, 27648, 768, 36, 768, 64, cis);
  conv3_relu_pool_k<<<blocks_for(64 * 16 * 9 * 17), 256, 0, stream>>>(
      cis, PW(37), PW(38), he1, 64, 1, 20, 36, 16);
  conv3_relu_pool_k<<<blocks_for(64 * 32 * 3 * 7), 256, 0, stream>>>(
      he1, PW(39), PW(40), he2, 64, 16, 9, 17, 32);
  gemm(stream, AM_PLAIN, ACT_RELU, he2, PW(41), PW(42), ent_fc, 64, 768, 672, 672);
  gemm(stream, AM_PLAIN, ACT_GELU, ent_fc, PW(43), PW(44), fc2_h, 32, 1536, 1536, 1536);
  layernorm_k<<<32, 256, 0, stream>>>(fc2_h, PW(45), PW(46), fc2_ln, 1536);
  gemm(stream, AM_PLAIN, ACT_NONE, fc2_ln, PW(47), PW(48), logit2, 32, 768, 1536, 1536);

  // ---- img self branch -> logit3 ----
  normdot_k<<<blocks_for((long long)32 * 36 * 36 * 32), 256, 0, stream>>>(
      img, 55296, 1536, 36, img + 384, 55296, 1536, 36, 384, 32, cii);
  conv3_relu_pool_k<<<blocks_for(32 * 16 * 17 * 17), 256, 0, stream>>>(
      cii, PW(49), PW(50), hi1, 32, 1, 36, 36, 16);
  conv3_relu_pool_k<<<blocks_for(32 * 32 * 7 * 7), 256, 0, stream>>>(
      hi1, PW(51), PW(52), hi2, 32, 16, 17, 17, 32);
  gemm(stream, AM_PLAIN, ACT_RELU, hi2, PW(53), PW(54), img_fc, 32, 768, 1568, 1568);
  gemm(stream, AM_PLAIN, ACT_GELU, img_fc, PW(55), PW(56), fc3_h, 32, 768, 768, 768);
  layernorm_k<<<32, 256, 0, stream>>>(fc3_h, PW(57), PW(58), logit3, 768);

  // ---- fusion head ----
  concat4_k<<<blocks_for(32 * 3072), 256, 0, stream>>>(logit1, logit2, logit3, logit4, clb);
  gemm(stream, AM_PLAIN, ACT_GELU, clb, PW(59), PW(60), f_h, 32, 3072, 3072, 3072);
  layernorm_k<<<32, 256, 0, stream>>>(f_h, PW(61), PW(62), f_ln, 3072);
  gemm(stream, AM_PLAIN, ACT_RELU, f_ln, PW(63), PW(64), f2, 32, 768, 3072, 3072);
  gemm(stream, AM_PLAIN, ACT_RELU, f2, PW(65), PW(66), f3, 32, 192, 768, 768);
  gemm(stream, AM_PLAIN, ACT_NONE, f3, PW(67), PW(68), (float*)d_out, 32, 2, 192, 192);
#undef PW
}
